// Top1Gate_20478404067792
// MI455X (gfx1250) — matrix operand order repaired and checked by reading, not compiled, not run on hardware
//
#include <hip/hip_runtime.h>
#include <hip/hip_bf16.h>

typedef __attribute__((ext_vector_type(16))) __bf16       v16bf;
typedef __attribute__((ext_vector_type(8)))  unsigned int v8u;
typedef __attribute__((ext_vector_type(8)))  float        v8f;

#define BATCH 16384
#define DM    2048
#define NE    64

#define THREADS        256    // 8 waves of 32
#define WAVES_PER_BLK  8
#define ROWS_PER_WAVE  16
#define KCHUNK         32     // K per bf16 WMMA
#define WIN_CHUNKS     8      // K-chunks staged per LDS window (256 K)
#define NWIN           (DM / (KCHUNK * WIN_CHUNKS))   // 8

// Two-term bf16 truncation split of 16 floats, packed with v_perm_b32.
// hi = top 16 bits of f (exact truncation), lo = bf16(f - hi) (truncated).
// 3 VALU per element: v_and, v_sub_f32, and half a v_perm each for hi/lo.
__device__ __forceinline__ void split16(const float* v, v8u& hi, v8u& lo) {
#pragma unroll
    for (int i = 0; i < 8; ++i) {
        float f0 = v[2 * i], f1 = v[2 * i + 1];
        float h0 = __uint_as_float(__float_as_uint(f0) & 0xffff0000u);
        float h1 = __uint_as_float(__float_as_uint(f1) & 0xffff0000u);
        float l0 = f0 - h0;            // exact: next 16 mantissa bits
        float l1 = f1 - h1;
        // pack {f1.hi16, f0.hi16} into one dword
        hi[i] = __builtin_amdgcn_perm(__float_as_uint(f1), __float_as_uint(f0),
                                      0x07060302u);
        lo[i] = __builtin_amdgcn_perm(__float_as_uint(l1), __float_as_uint(l0),
                                      0x07060302u);
    }
}

__global__ __launch_bounds__(THREADS) void top1gate_kernel(
    const float* __restrict__ x, const float* __restrict__ w,
    float* __restrict__ idx_out, float* __restrict__ score_out,
    float* __restrict__ mask_out)
{
    // hi frags: [0..1023], lo frags: [1024..2047]; each entry = 16 bf16 = 32 B. 64 KB.
    __shared__ v8u smem[2 * WIN_CHUNKS * 4 * 32];

    const int tid  = threadIdx.x;
    const int wave = tid >> 5;
    const int lane = tid & 31;
    const int half = lane >> 4;     // 0: lanes 0-15, 1: lanes 16-31
    const int lq   = lane & 15;

    const int mbase = (blockIdx.x * WAVES_PER_BLK + wave) * ROWS_PER_WAVE;

    v8f acc[4] = {{0,0,0,0,0,0,0,0}, {0,0,0,0,0,0,0,0},
                  {0,0,0,0,0,0,0,0}, {0,0,0,0,0,0,0,0}};

    // A-fragment addressing: lane holds row mbase+lq, K sub-offset half*8
    const float* xrow = x + (size_t)(mbase + lq) * DM + half * 8;

    for (int win = 0; win < NWIN; ++win) {
        __syncthreads();
        // ---- cooperative split of w_gate into B-fragment layout for this window ----
        // 8 chunks * 4 tiles * 32 lanes = 1024 slots; 4 slots per thread.
#pragma unroll
        for (int j = 0; j < 4; ++j) {
            int slot = j * THREADS + tid;       // 0..1023
            int s  = slot >> 7;                 // chunk within window
            int rm = slot & 127;
            int t  = rm >> 5;                   // expert tile
            int l  = rm & 31;                   // consuming lane
            int e  = t * 16 + (l & 15);         // expert (B column)
            int kk = win * (WIN_CHUNKS * KCHUNK) + s * KCHUNK + (l >> 4) * 8;
            const float4* wp = (const float4*)(w + (size_t)e * DM + kk);
            float4 f0 = wp[0], f1 = wp[1], f2 = wp[4], f3 = wp[5];
            float f[16] = { f0.x, f0.y, f0.z, f0.w,  f1.x, f1.y, f1.z, f1.w,
                            f2.x, f2.y, f2.z, f2.w,  f3.x, f3.y, f3.z, f3.w };
            v8u hi, lo;
            split16(f, hi, lo);
            int idx = (s * 4 + t) * 32 + l;
            smem[idx]        = hi;
            smem[idx + 1024] = lo;
        }
        __syncthreads();

        // ---- compute the window: per chunk load/split A, 12 WMMAs ----
        for (int s = 0; s < WIN_CHUNKS; ++s) {
            int k0 = win * (WIN_CHUNKS * KCHUNK) + s * KCHUNK;
            const float4* xp = (const float4*)(xrow + k0);
            // A 16-bit layout: elems 0-7 = K(koff+0..7), elems 8-15 = K(koff+16..23)
            float4 a0 = xp[0], a1 = xp[1], a2 = xp[4], a3 = xp[5];
            float a[16] = { a0.x, a0.y, a0.z, a0.w,  a1.x, a1.y, a1.z, a1.w,
                            a2.x, a2.y, a2.z, a2.w,  a3.x, a3.y, a3.z, a3.w };
            v8u ahu, alu;
            split16(a, ahu, alu);
            v16bf ah = __builtin_bit_cast(v16bf, ahu);
            v16bf al = __builtin_bit_cast(v16bf, alu);
#pragma unroll
            for (int t = 0; t < 4; ++t) {
                int idx = (s * 4 + t) * 32 + lane;
                v16bf bh = __builtin_bit_cast(v16bf, smem[idx]);
                v16bf bl = __builtin_bit_cast(v16bf, smem[idx + 1024]);
                acc[t] = __builtin_amdgcn_wmma_f32_16x16x32_bf16(
                             false, ah, false, bh, (short)0, acc[t], false, false);
                acc[t] = __builtin_amdgcn_wmma_f32_16x16x32_bf16(
                             false, al, false, bh, (short)0, acc[t], false, false);
                acc[t] = __builtin_amdgcn_wmma_f32_16x16x32_bf16(
                             false, ah, false, bl, (short)0, acc[t], false, false);
            }
        }
    }

    // ---- epilogue: per-row argmax over 64 experts ----
    // C layout: vgpr r, lane l -> row = r + 8*(l>=16), expert = 16t + l%16
#pragma unroll
    for (int r = 0; r < 8; ++r) {
        float bv = acc[0][r];
        int   bi = lq;
#pragma unroll
        for (int t = 1; t < 4; ++t) {
            float v = acc[t][r];
            int   e = t * 16 + lq;
            if (v > bv) { bv = v; bi = e; }    // strict > keeps first occurrence
        }
#pragma unroll
        for (int m = 1; m < 16; m <<= 1) {     // stays within the 16-lane half
            float ov = __shfl_xor(bv, m, 32);
            int   oi = __shfl_xor(bi, m, 32);
            if (ov > bv || (ov == bv && oi < bi)) { bv = ov; bi = oi; }
        }
        int row = mbase + r + half * 8;
        if (lq == 0) {
            idx_out[row]   = (float)bi;
            score_out[row] = bv;
        }
        int e0 = lq * 4;
        float4 mk;
        mk.x = (e0 + 0 == bi) ? 1.f : 0.f;
        mk.y = (e0 + 1 == bi) ? 1.f : 0.f;
        mk.z = (e0 + 2 == bi) ? 1.f : 0.f;
        mk.w = (e0 + 3 == bi) ? 1.f : 0.f;
        *(float4*)(mask_out + (size_t)row * NE + e0) = mk;
    }
}

extern "C" void kernel_launch(void* const* d_in, const int* in_sizes, int n_in,
                              void* d_out, int out_size, void* d_ws, size_t ws_size,
                              hipStream_t stream) {
    const float* x = (const float*)d_in[0];
    const float* w = (const float*)d_in[1];
    float* out   = (float*)d_out;
    float* idx   = out;                 // [BATCH]   (argmax, stored as f32)
    float* score = out + BATCH;         // [BATCH]
    float* mask  = out + 2 * BATCH;     // [BATCH*NE]
    dim3 grid(BATCH / (ROWS_PER_WAVE * WAVES_PER_BLK));   // 128 blocks
    top1gate_kernel<<<grid, THREADS, 0, stream>>>(x, w, idx, score, mask);
}
